// DifferentiableHMM_Centered_33303176413547
// MI455X (gfx1250) — 8-bit matrix operand order repaired and checked by reading, compiled, not run on hardware
//
#include <hip/hip_runtime.h>
#include <stdint.h>
#include <math.h>


#define NSTATE 3
#define TILE 1024
#define THREADS 256
#define TAU_INV 10.0f
#define HALF_LOG_2PI 0.9189385332046727f
#define GROUPS_PER_BLOCK 8

typedef unsigned int v4u __attribute__((ext_vector_type(4)));
typedef int          v4i __attribute__((ext_vector_type(4)));
typedef int          v8i __attribute__((ext_vector_type(8)));

#if defined(__gfx1250__)
#if __has_builtin(__builtin_amdgcn_tensor_load_to_lds) && __has_builtin(__builtin_amdgcn_s_wait_tensorcnt)
#define USE_TDM 1
#endif
#if __has_builtin(__builtin_amdgcn_wmma_i32_16x16x64_iu8)
#define USE_WMMA 1
#endif
#endif

__device__ __forceinline__ int pick_state(float xval, float g0v, float g1v, float g2v,
                                          float m0, float m2,
                                          float is0, float is1, float is2,
                                          float c0, float c1, float c2) {
  const float z0 = (xval - m0) * is0;
  const float z1 = xval * is1;
  const float z2 = (xval - m2) * is2;
  const float l0 = (-0.5f * z0 * z0 + c0 + g0v) * TAU_INV;
  const float l1 = (-0.5f * z1 * z1 + c1 + g1v) * TAU_INV;
  const float l2 = (-0.5f * z2 * z2 + c2 + g2v) * TAU_INV;
  int idx = 0; float best = l0;
  if (l1 > best) { best = l1; idx = 1; }
  if (l2 > best) { idx = 2; }
  return idx;
}

__global__ __launch_bounds__(THREADS)
void hmm_forward_kernel(const float* __restrict__ x,
                        const float* __restrict__ gn,
                        const float* __restrict__ means2,
                        const float* __restrict__ lstds,
                        float* __restrict__ nc,
                        unsigned char* __restrict__ st,
                        int B, int pitch,
                        long long total) {
  __shared__ float gtile[TILE * NSTATE];

  const long long base = (long long)blockIdx.x * TILE;
  const int t = threadIdx.x;

  const float m0 = means2[0];
  const float m2 = means2[1];
  const float sd0 = expf(lstds[0]) + 1e-6f;
  const float sd1 = expf(lstds[1]) + 1e-6f;
  const float sd2 = expf(lstds[2]) + 1e-6f;
  const float is0 = 1.0f / sd0, is1 = 1.0f / sd1, is2 = 1.0f / sd2;
  const float c0 = -logf(sd0) - HALF_LOG_2PI;
  const float c1 = -logf(sd1) - HALF_LOG_2PI;
  const float c2 = -logf(sd2) - HALF_LOG_2PI;

  const bool full = (base + TILE) <= total;

  if (full) {
#if defined(USE_TDM)
    if (t < 32) {
      const unsigned lds_off = (unsigned)(unsigned long long)&gtile[0];
      const unsigned long long ga =
          (unsigned long long)(const void*)(gn + base * NSTATE);
      const unsigned n = TILE * NSTATE;

      v4u g0;
      g0.x = 1u;
      g0.y = lds_off;
      g0.z = (unsigned)(ga & 0xFFFFFFFFu);
      g0.w = (unsigned)((ga >> 32) & 0x01FFFFFFu) | (2u << 30);

      v8i g1 = {
        (int)0x00020000u,
        (int)((n & 0xFFFFu) << 16),
        (int)(((n >> 16) & 0xFFFFu) | (1u << 16)),
        (int)((n & 0xFFFFu) << 16),
        0,
        (int)n,
        0, 0
      };
      v4i gz = {0, 0, 0, 0};
#if __clang_major__ >= 23
      v8i gz8 = {0, 0, 0, 0, 0, 0, 0, 0};
      __builtin_amdgcn_tensor_load_to_lds(g0, g1, gz, gz, gz8, 0);
#else
      __builtin_amdgcn_tensor_load_to_lds(g0, g1, gz, gz, 0);
#endif
      __builtin_amdgcn_s_wait_tensorcnt(0);
    }
#else
    const float4* gsrc = (const float4*)(gn + base * NSTATE);
    for (int i = t; i < (TILE * NSTATE) / 4; i += THREADS)
      ((float4*)gtile)[i] = gsrc[i];
#endif

    const float4 xv = *(const float4*)(x + base + 4 * t);
    __syncthreads();

    const float xs[4] = {xv.x, xv.y, xv.z, xv.w};
    float res[4];
    int   stv[4];
#pragma unroll
    for (int j = 0; j < 4; ++j) {
      const int e = 4 * t + j;
      const int idx = pick_state(xs[j], gtile[e*3+0], gtile[e*3+1], gtile[e*3+2],
                                 m0, m2, is0, is1, is2, c0, c1, c2);
      stv[j] = idx;
      res[j] = (idx == 0) ? m0 : ((idx == 1) ? 0.0f : m2);
    }
    *(float4*)(nc + base + 4 * t) = make_float4(res[0], res[1], res[2], res[3]);

    if (st) {
      const unsigned e0 = (unsigned)(base + 4 * t);
      const unsigned s  = e0 / (unsigned)B;
      const unsigned b  = e0 - s * (unsigned)B;
      unsigned h[12];
#pragma unroll
      for (int j = 0; j < 4; ++j) {
        h[3*j+0] = (stv[j] == 0);
        h[3*j+1] = (stv[j] == 1);
        h[3*j+2] = (stv[j] == 2);
      }
      unsigned int* p = (unsigned int*)(st + (size_t)s * pitch + 3u * b);
      p[0] = h[0] | (h[1] << 8) | (h[2]  << 16) | (h[3]  << 24);
      p[1] = h[4] | (h[5] << 8) | (h[6]  << 16) | (h[7]  << 24);
      p[2] = h[8] | (h[9] << 8) | (h[10] << 16) | (h[11] << 24);
    }
  } else {
    for (int j = 0; j < 4; ++j) {
      const long long e = base + 4 * t + j;
      if (e >= total) break;
      const int idx = pick_state(x[e], gn[e*3+0], gn[e*3+1], gn[e*3+2],
                                 m0, m2, is0, is1, is2, c0, c1, c2);
      nc[e] = (idx == 0) ? m0 : ((idx == 1) ? 0.0f : m2);
      if (st) {
        const unsigned s = (unsigned)(e / B), b = (unsigned)(e - (long long)s * B);
        unsigned char* q = st + (size_t)s * pitch + 3u * b;
        q[0] = (idx == 0); q[1] = (idx == 1); q[2] = (idx == 2);
      }
    }
  }
}

__global__ __launch_bounds__(256)
void smooth_wmma_kernel(const unsigned char* __restrict__ st,
                        const int* __restrict__ edges,
                        int E, int B, int pitch, int ngroups,
                        unsigned int* __restrict__ counter) {
#if defined(USE_WMMA)
  const int lane = threadIdx.x & 31;
  const int wave = threadIdx.x >> 5;
  const int g = blockIdx.x * GROUPS_PER_BLOCK + wave;

  __shared__ unsigned int acc;
  if (threadIdx.x == 0) acc = 0u;
  __syncthreads();

  if (g < ngroups) {
    const int eidx = g * 16 + (lane & 15);
    const unsigned char* pa = st + (size_t)edges[eidx]     * pitch + ((lane & 16) ? 16 : 0);
    const unsigned char* pb = st + (size_t)edges[E + eidx] * pitch + ((lane & 16) ? 16 : 0);

    v8i c = {0, 0, 0, 0, 0, 0, 0, 0};
    const int nk = pitch >> 6;
    for (int k = 0; k < nk; ++k) {
      const unsigned char* qa = pa + (k << 6);
      const unsigned char* qb = pb + (k << 6);
      const uint2 a0 = *(const uint2*)(qa);
      const uint2 a1 = *(const uint2*)(qa + 8);
      const uint2 a2 = *(const uint2*)(qa + 32);
      const uint2 a3 = *(const uint2*)(qa + 40);
      const uint4 b0 = *(const uint4*)(qb);
      const uint4 b1 = *(const uint4*)(qb + 32);
      const v8i av = {(int)a0.x,(int)a0.y,(int)a1.x,(int)a1.y,
                      (int)a2.x,(int)a2.y,(int)a3.x,(int)a3.y};
      const v8i bv = {(int)b0.x,(int)b0.y,(int)b0.z,(int)b0.w,
                      (int)b1.x,(int)b1.y,(int)b1.z,(int)b1.w};
      c = __builtin_amdgcn_wmma_i32_16x16x64_iu8(false, av, false, bv, c,
                                                 false, false);
    }
    if (lane < 8 || lane >= 24) {
      const int li = lane & 7;
      int d = c[0];
      if (li == 1) d = c[1];
      if (li == 2) d = c[2];
      if (li == 3) d = c[3];
      if (li == 4) d = c[4];
      if (li == 5) d = c[5];
      if (li == 6) d = c[6];
      if (li == 7) d = c[7];
      atomicAdd(&acc, (unsigned)(B - d));
    }
  }
  __syncthreads();
  if (threadIdx.x == 0) atomicAdd(counter, acc);
#else
  (void)st; (void)edges; (void)E; (void)B; (void)pitch; (void)ngroups; (void)counter;
#endif
}

__global__ __launch_bounds__(256)
void smooth_count_kernel(const float* __restrict__ nc,
                         const int* __restrict__ edges,
                         int E, int B, int eoff,
                         unsigned int* __restrict__ counter) {
  const int e = eoff + blockIdx.x;
  const long long row = edges[e];
  const long long col = edges[E + e];
  const float4* a = (const float4*)(nc + row * B);
  const float4* b = (const float4*)(nc + col * B);
  const int n4 = B >> 2;

  unsigned int c = 0;
  for (int i = threadIdx.x; i < n4; i += blockDim.x) {
    const float4 av = a[i], bv = b[i];
    c += (unsigned)(av.x != bv.x) + (unsigned)(av.y != bv.y) +
         (unsigned)(av.z != bv.z) + (unsigned)(av.w != bv.w);
  }
  for (int i = (n4 << 2) + threadIdx.x; i < B; i += blockDim.x)
    c += (unsigned)(nc[row * B + i] != nc[col * B + i]);

  __shared__ unsigned int sred[256];
  sred[threadIdx.x] = c;
  __syncthreads();
  for (int s = 128; s > 0; s >>= 1) {
    if ((int)threadIdx.x < s) sred[threadIdx.x] += sred[threadIdx.x + s];
    __syncthreads();
  }
  if (threadIdx.x == 0) atomicAdd(counter, sred[0]);
}

__global__ void init_counter_kernel(unsigned int* counter) { *counter = 0u; }

__global__ void pad_zero_kernel(unsigned char* __restrict__ st,
                                int S, int B, int pitch) {
  const int padw = (pitch - 3 * B) >> 2;
  const long long i = (long long)blockIdx.x * blockDim.x + threadIdx.x;
  if (i < (long long)S * padw) {
    const int s = (int)(i / padw), o = (int)(i % padw);
    *(unsigned int*)(st + (size_t)s * pitch + 3 * B + 4 * o) = 0u;
  }
}

__global__ void finalize_kernel(const unsigned int* __restrict__ counter,
                                float* __restrict__ out_scalar,
                                int E, int B) {
  const double sum = 2.0 * (double)(*counter);
  const double denom = (double)E * (double)B * 3.0;
  *out_scalar = (float)(0.1 * sum / denom);
}

extern "C" void kernel_launch(void* const* d_in, const int* in_sizes, int n_in,
                              void* d_out, int out_size, void* d_ws, size_t ws_size,
                              hipStream_t stream) {
  const float* x     = (const float*)d_in[0];
  const int*   edges = (const int*)d_in[2];
  const float* gn    = (const float*)d_in[3];
  const float* means = (const float*)d_in[4];
  const float* lstds = (const float*)d_in[5];
  (void)n_in; (void)out_size;

  const int B  = in_sizes[1];
  const long long SB = in_sizes[0];
  const int S  = (int)(SB / B);
  const int E  = in_sizes[2] / 2;

  float* nc = (float*)d_out;
  unsigned int* counter = (unsigned int*)d_ws;

  const int pitch = ((3 * B + 63) / 64) * 64;
  const size_t need = 64 + (size_t)S * pitch;
  unsigned char* st = (ws_size >= need) ? ((unsigned char*)d_ws + 64) : nullptr;

  init_counter_kernel<<<1, 1, 0, stream>>>(counter);
  if (st) {
    const long long padTot = (long long)S * ((pitch - 3 * B) >> 2);
    pad_zero_kernel<<<(int)((padTot + 255) / 256), 256, 0, stream>>>(st, S, B, pitch);
  }

  const long long nblocks = (SB + TILE - 1) / TILE;
  hmm_forward_kernel<<<(int)nblocks, THREADS, 0, stream>>>(
      x, gn, means, lstds, nc, st, B, pitch, SB);

  if (st) {
    const int ngroups = E / 16;
    if (ngroups > 0) {
      const int gblocks = (ngroups + GROUPS_PER_BLOCK - 1) / GROUPS_PER_BLOCK;
      smooth_wmma_kernel<<<gblocks, 256, 0, stream>>>(st, edges, E, B, pitch,
                                                      ngroups, counter);
    }
    const int rem = E - ngroups * 16;
    if (rem > 0)
      smooth_count_kernel<<<rem, 256, 0, stream>>>(nc, edges, E, B,
                                                   ngroups * 16, counter);
  } else {
    smooth_count_kernel<<<E, 256, 0, stream>>>(nc, edges, E, B, 0, counter);
  }

  finalize_kernel<<<1, 1, 0, stream>>>(counter, nc + SB, E, B);
}
